// MHSA2D_47974784696755
// MI455X (gfx1250) — matrix operand order repaired and checked by reading, not compiled, not run on hardware
//
#include <hip/hip_runtime.h>

typedef _Float16 v16h __attribute__((ext_vector_type(16)));
typedef _Float16 v8h  __attribute__((ext_vector_type(8)));
typedef _Float16 v4h  __attribute__((ext_vector_type(4)));
typedef float    v8f  __attribute__((ext_vector_type(8)));
typedef float    v4f  __attribute__((ext_vector_type(4)));

#define WMMA_F32_F16(a, b, c) \
  __builtin_amdgcn_wmma_f32_16x16x32_f16(false, (a), false, (b), (short)0, (c), false, false)

#define B_      2
#define C_      256
#define N_      4096     // H*W
#define HEADS_  8
#define DH_     64
#define INNER_  512
#define QKVROWS 1536

// ---------------------------------------------------------------------------
// WMMA fragment loaders (CDNA5 16x16x32 f16 layouts, CDNA5 ISA 7.12.2)
// A (16xK tile, K-chunk of 32): lane<16 -> row=lane, K = {0..7, 16..23}
//                               lane>=16 -> row=lane-16, K = {8..15, 24..31}
// B (32x16 tile):               lane<16 -> col=lane, K = {0..7, 16..23}
//                               lane>=16 -> col=lane-16, K = {8..15, 24..31}
// C/D (16x16 f32):              elem r -> row = r + (lane<16 ? 0 : 8), col = lane&15
// ---------------------------------------------------------------------------
__device__ __forceinline__ v16h wmma_load_a(const _Float16* base, int ld) {
  const int lane = threadIdx.x & 31;
  const int m  = lane & 15;
  const int kb = (lane < 16) ? 0 : 8;
  const _Float16* p = base + m * ld + kb;
  const v8h lo = *(const v8h*)(p);        // K = kb .. kb+7
  const v8h hi = *(const v8h*)(p + 16);   // K = 16+kb .. 16+kb+7
  v16h r;
#pragma unroll
  for (int i = 0; i < 8; ++i) { r[i] = lo[i]; r[i + 8] = hi[i]; }
  return r;
}

__device__ __forceinline__ v16h wmma_load_b(const _Float16* base, int ld) {
  const int lane = threadIdx.x & 31;
  const int n  = lane & 15;
  const int kb = (lane < 16) ? 0 : 8;
  const _Float16* p = base + n * ld + kb;
  const v8h lo = *(const v8h*)(p);        // K = kb .. kb+7
  const v8h hi = *(const v8h*)(p + 16);   // K = 16+kb .. 16+kb+7
  v16h r;
#pragma unroll
  for (int i = 0; i < 8; ++i) { r[i] = lo[i]; r[i + 8] = hi[i]; }
  return r;
}

// Reductions across the 16-lane half-group (matches C-fragment row striping).
__device__ __forceinline__ float row_max16(float v) {
#pragma unroll
  for (int off = 8; off >= 1; off >>= 1) v = fmaxf(v, __shfl_xor(v, off, 32));
  return v;
}
__device__ __forceinline__ float row_sum16(float v) {
#pragma unroll
  for (int off = 8; off >= 1; off >>= 1) v += __shfl_xor(v, off, 32);
  return v;
}

__device__ __forceinline__ void wait_asynccnt0() {
#if __has_builtin(__builtin_amdgcn_s_wait_asynccnt)
  __builtin_amdgcn_s_wait_asynccnt(0);
#else
  asm volatile("s_wait_asynccnt 0x0" ::: "memory");
#endif
}

// ---------------------------------------------------------------------------
// Kernel 1: QKV projection.  out[o,n] = sum_c w_qkv[o,c] * x[b,c,n]
// 128x128 tile per WG, 8 waves (4x2), K-step 32, f32->f16 staged in LDS.
// Epilogue packs 8 contiguous d-values -> one b128 store into Q/K/V[b][h][n][d];
// Q pre-scaled by 1/sqrt(64).
// ---------------------------------------------------------------------------
#define G_TM 128
#define G_TN 128
#define G_TK 32
#define G_LDA 40   // (TK+8) halves: 80B rows keep 16B alignment, break bank strides
#define G_LDB 40

__global__ void __launch_bounds__(256)
qkv_proj_wmma(const float* __restrict__ x, const float* __restrict__ w_qkv,
              _Float16* __restrict__ Qd, _Float16* __restrict__ Kd,
              _Float16* __restrict__ Vd) {
  __shared__ __align__(16) _Float16 As[G_TM * G_LDA];
  __shared__ __align__(16) _Float16 Bs[G_TN * G_LDB];   // stored [n][k]

  const int tid  = threadIdx.x;
  const int lane = tid & 31;
  const int wave = tid >> 5;
  const int wm   = wave >> 1;           // 0..3 -> 32-row strip
  const int wn   = wave & 1;            // 0..1 -> 64-col strip
  const int m0   = blockIdx.x * G_TM;
  const int n0   = blockIdx.y * G_TN;
  const int b    = blockIdx.z;
  const float* X = x + (size_t)b * C_ * N_;

  v8f zero = {};
  v8f acc[2][4];
#pragma unroll
  for (int i = 0; i < 2; ++i)
#pragma unroll
    for (int j = 0; j < 4; ++j) acc[i][j] = zero;

  for (int k0 = 0; k0 < C_; k0 += G_TK) {
    __syncthreads();
    // A tile: 128x32, vectorized float4 load -> v4h store (4 chunks/thread)
#pragma unroll
    for (int it = 0; it < 4; ++it) {
      const int q  = tid + it * 256;
      const int r  = q >> 3;            // 0..127
      const int c4 = (q & 7) * 4;       // 0..28
      const v4f w4 = *(const v4f*)&w_qkv[(size_t)(m0 + r) * C_ + k0 + c4];
      v4h h4;
#pragma unroll
      for (int u = 0; u < 4; ++u) h4[u] = (_Float16)w4[u];
      *(v4h*)&As[r * G_LDA + c4] = h4;
    }
    // B tile: 32x128, float4 along n, transposed b16 stores into [n][k]
#pragma unroll
    for (int it = 0; it < 4; ++it) {
      const int q  = tid + it * 256;
      const int kk = q >> 5;            // 0..31
      const int n4 = (q & 31) * 4;      // 0..124
      const v4f x4 = *(const v4f*)&X[(size_t)(k0 + kk) * N_ + n0 + n4];
#pragma unroll
      for (int u = 0; u < 4; ++u) Bs[(n4 + u) * G_LDB + kk] = (_Float16)x4[u];
    }
    __syncthreads();

    const v16h a0 = wmma_load_a(&As[(wm * 32 +  0) * G_LDA], G_LDA);
    const v16h a1 = wmma_load_a(&As[(wm * 32 + 16) * G_LDA], G_LDA);
    v16h bf[4];
#pragma unroll
    for (int j = 0; j < 4; ++j)
      bf[j] = wmma_load_b(&Bs[(wn * 64 + j * 16) * G_LDB], G_LDB);
#pragma unroll
    for (int j = 0; j < 4; ++j) {
      acc[0][j] = WMMA_F32_F16(a0, bf[j], acc[0][j]);
      acc[1][j] = WMMA_F32_F16(a1, bf[j], acc[1][j]);
    }
  }

  const int nloc = lane & 15;
  const int mofs = (lane < 16) ? 0 : 8;
#pragma unroll
  for (int i = 0; i < 2; ++i)
#pragma unroll
    for (int j = 0; j < 4; ++j) {
      const int obase = m0 + wm * 32 + i * 16 + mofs;   // multiple of 8
      const int n     = n0 + wn * 64 + j * 16 + nloc;   // pixel
      const int which = obase >> 9;                      // 0=q 1=k 2=v (uniform/tile)
      const int h     = (obase >> 6) & 7;
      const int d0    = obase & 63;                      // multiple of 8
      const float sc  = (which == 0) ? 0.125f : 1.0f;    // fold softmax scale into Q
      v8h pack;
#pragma unroll
      for (int r = 0; r < 8; ++r) pack[r] = (_Float16)(acc[i][j][r] * sc);
      _Float16* dst = (which == 0) ? Qd : ((which == 1) ? Kd : Vd);
      *(v8h*)&dst[(((size_t)b * HEADS_ + h) * N_ + n) * DH_ + d0] = pack;
    }
}

// ---------------------------------------------------------------------------
// Kernel 2: flash attention.  One WG per (128 q-rows, b*8+h); each wave owns
// 16 q-rows.  K tile streamed global->LDS via GLOBAL_LOAD_ASYNC_TO_LDS_B128
// (ASYNCcnt); V transposed manually; online softmax; P goes through a per-wave
// LDS patch to convert C-fragment layout -> A-fragment layout.
// ---------------------------------------------------------------------------
#define F_BK  32
#define F_KLD 72    // Klds[key][d]   (64+8) halves -> 144B rows (16B multiple)
#define F_VLD 40    // Vt[d][key]     (32+8)
#define F_PLD 40    // Pw[row][key]   (32+8)

__global__ void __launch_bounds__(256)
flash_attn_wmma(const _Float16* __restrict__ Qd, const _Float16* __restrict__ Kd,
                const _Float16* __restrict__ Vd, _Float16* __restrict__ att) {
  __shared__ __align__(16) _Float16 Klds[F_BK * F_KLD];
  __shared__ __align__(16) _Float16 Vt[DH_ * F_VLD];
  __shared__ __align__(16) _Float16 Plds[8 * 16 * F_PLD];

  const int tid   = threadIdx.x;
  const int lane  = tid & 31;
  const int wave  = tid >> 5;
  const int bh    = blockIdx.y;                   // b*8 + h
  const int qbase = blockIdx.x * 128 + wave * 16;

  const _Float16* Qp = Qd + ((size_t)bh * N_ + qbase) * DH_;
  const _Float16* Kp = Kd + (size_t)bh * N_ * DH_;
  const _Float16* Vp = Vd + (size_t)bh * N_ * DH_;

  // Q fragments: 16x64 = two 16x32 A-frags, loaded once from global.
  v16h qf[2];
#pragma unroll
  for (int c = 0; c < 2; ++c) qf[c] = wmma_load_a(Qp + c * 32, DH_);

  v8f zero = {};
  v8f oacc[4];
#pragma unroll
  for (int t = 0; t < 4; ++t) oacc[t] = zero;
  float mstate[8], lstate[8];
#pragma unroll
  for (int r = 0; r < 8; ++r) { mstate[r] = -3.0e38f; lstate[r] = 0.0f; }

  _Float16* Pw = &Plds[wave * 16 * F_PLD];

  // per-thread tile-copy coordinates (32 rows x 8 chunks of 16B)
  const int kk_cp = tid >> 3;          // 0..31 key row
  const int c8_cp = (tid & 7) * 8;     // half-offset within row (16B chunks)

  for (int kb0 = 0; kb0 < N_; kb0 += F_BK) {
    __syncthreads();
    // ---- K tile: 256 async b128 copies, tracked with ASYNCcnt -------------
    {
      const _Float16* gsrc = Kp + (size_t)(kb0 + kk_cp) * DH_ + c8_cp;
      const unsigned ldsoff = (unsigned)(size_t)(const void*)&Klds[kk_cp * F_KLD + c8_cp];
      const unsigned long long ga = (unsigned long long)(const void*)gsrc;
      asm volatile("global_load_async_to_lds_b128 %0, %1, off"
                   :: "v"(ldsoff), "v"(ga) : "memory");
    }
    // ---- V tile transposed: one b128 load + 8 b16 LDS stores per thread ---
    {
      const v8h vv = *(const v8h*)(Vp + (size_t)(kb0 + kk_cp) * DH_ + c8_cp);
#pragma unroll
      for (int u = 0; u < 8; ++u) Vt[(c8_cp + u) * F_VLD + kk_cp] = vv[u];
    }
    wait_asynccnt0();
    __syncthreads();
    if (kb0 + F_BK < N_) {   // uniform branch: EXEC stays all-ones
      __builtin_prefetch(Kp + (size_t)(kb0 + F_BK) * DH_ + lane * 64, 0, 1);
      __builtin_prefetch(Vp + (size_t)(kb0 + F_BK) * DH_ + lane * 64, 0, 1);
    }

    // S = (Q*scale) @ K^T : two 16x16 tiles, each accumulating over d=64
    v16h kf[2][2];
#pragma unroll
    for (int j2 = 0; j2 < 2; ++j2)
#pragma unroll
      for (int c = 0; c < 2; ++c)
        kf[j2][c] = wmma_load_b(&Klds[(j2 * 16) * F_KLD + c * 32], F_KLD);

    v8f s[2];
#pragma unroll
    for (int j2 = 0; j2 < 2; ++j2) {
      v8f sj = zero;
      sj = WMMA_F32_F16(qf[0], kf[j2][0], sj);
      sj = WMMA_F32_F16(qf[1], kf[j2][1], sj);
      s[j2] = sj;
    }

    // online softmax per row (rows striped r + 8*(lane>=16), cols = lane&15)
#pragma unroll
    for (int r = 0; r < 8; ++r) {
      const float mx  = row_max16(fmaxf(s[0][r], s[1][r]));
      const float mn  = fmaxf(mstate[r], mx);
      const float cor = __expf(mstate[r] - mn);
      const float p0  = __expf(s[0][r] - mn);
      const float p1  = __expf(s[1][r] - mn);
      lstate[r] = lstate[r] * cor + row_sum16(p0 + p1);
      mstate[r] = mn;
#pragma unroll
      for (int t = 0; t < 4; ++t) oacc[t][r] *= cor;
      const int prow = r + ((lane < 16) ? 0 : 8);
      Pw[prow * F_PLD + (lane & 15)]      = (_Float16)p0;
      Pw[prow * F_PLD + 16 + (lane & 15)] = (_Float16)p1;
    }

    // O += P(16x32) @ V(32x64)   (same-wave LDS ops are in-order: no barrier)
    const v16h pf = wmma_load_a(Pw, F_PLD);
    v16h vf[4];
#pragma unroll
    for (int t = 0; t < 4; ++t) vf[t] = wmma_load_b(&Vt[(t * 16) * F_VLD], F_VLD);
#pragma unroll
    for (int t = 0; t < 4; ++t) oacc[t] = WMMA_F32_F16(pf, vf[t], oacc[t]);
  }

  // epilogue: att[b][h*64+d][n] (f16); rows r are contiguous along n -> b128
  const int b = bh >> 3, h = bh & 7;
  float linv[8];
#pragma unroll
  for (int r = 0; r < 8; ++r) linv[r] = 1.0f / lstate[r];
  const int row0 = qbase + ((lane < 16) ? 0 : 8);
#pragma unroll
  for (int t = 0; t < 4; ++t) {
    const int inner = h * DH_ + t * 16 + (lane & 15);
    v8h pack;
#pragma unroll
    for (int r = 0; r < 8; ++r) pack[r] = (_Float16)(oacc[t][r] * linv[r]);
    *(v8h*)&att[((size_t)b * INNER_ + inner) * N_ + row0] = pack;
  }
}

// ---------------------------------------------------------------------------
// Kernel 3: output projection.  out[c,n] = sum_i w_out[c,i]*att[b,i,n] + b_out[c]
// ---------------------------------------------------------------------------
__global__ void __launch_bounds__(256)
out_proj_wmma(const float* __restrict__ w_out, const _Float16* __restrict__ att,
              const float* __restrict__ b_out, float* __restrict__ out) {
  __shared__ __align__(16) _Float16 As[G_TM * G_LDA];
  __shared__ __align__(16) _Float16 Bs[G_TN * G_LDB];

  const int tid  = threadIdx.x;
  const int lane = tid & 31;
  const int wave = tid >> 5;
  const int wm   = wave >> 1;
  const int wn   = wave & 1;
  const int m0   = blockIdx.x * G_TM;     // C_ = 256 -> grid.x = 2
  const int n0   = blockIdx.y * G_TN;
  const int b    = blockIdx.z;
  const _Float16* X = att + (size_t)b * INNER_ * N_;

  v8f zero = {};
  v8f acc[2][4];
#pragma unroll
  for (int i = 0; i < 2; ++i)
#pragma unroll
    for (int j = 0; j < 4; ++j) acc[i][j] = zero;

  for (int k0 = 0; k0 < INNER_; k0 += G_TK) {
    __syncthreads();
    // A tile: 128x32 f32 -> f16 (vectorized)
#pragma unroll
    for (int it = 0; it < 4; ++it) {
      const int q  = tid + it * 256;
      const int r  = q >> 3;
      const int c4 = (q & 7) * 4;
      const v4f w4 = *(const v4f*)&w_out[(size_t)(m0 + r) * INNER_ + k0 + c4];
      v4h h4;
#pragma unroll
      for (int u = 0; u < 4; ++u) h4[u] = (_Float16)w4[u];
      *(v4h*)&As[r * G_LDA + c4] = h4;
    }
    // B tile: 32x128 f16, v8h loads along n, transposed b16 stores
#pragma unroll
    for (int it = 0; it < 2; ++it) {
      const int q  = tid + it * 256;
      const int kk = q >> 4;            // 0..31
      const int n8 = (q & 15) * 8;      // 0..120
      const v8h x8 = *(const v8h*)&X[(size_t)(k0 + kk) * N_ + n0 + n8];
#pragma unroll
      for (int u = 0; u < 8; ++u) Bs[(n8 + u) * G_LDB + kk] = x8[u];
    }
    __syncthreads();

    const v16h a0 = wmma_load_a(&As[(wm * 32 +  0) * G_LDA], G_LDA);
    const v16h a1 = wmma_load_a(&As[(wm * 32 + 16) * G_LDA], G_LDA);
    v16h bf[4];
#pragma unroll
    for (int j = 0; j < 4; ++j)
      bf[j] = wmma_load_b(&Bs[(wn * 64 + j * 16) * G_LDB], G_LDB);
#pragma unroll
    for (int j = 0; j < 4; ++j) {
      acc[0][j] = WMMA_F32_F16(a0, bf[j], acc[0][j]);
      acc[1][j] = WMMA_F32_F16(a1, bf[j], acc[1][j]);
    }
  }

  const int nloc = lane & 15;
  const int mofs = (lane < 16) ? 0 : 8;
#pragma unroll
  for (int i = 0; i < 2; ++i)
#pragma unroll
    for (int j = 0; j < 4; ++j)
#pragma unroll
      for (int r = 0; r < 8; ++r) {
        const int cc = m0 + wm * 32 + i * 16 + r + mofs;
        const int n  = n0 + wn * 64 + j * 16 + nloc;
        out[((size_t)b * C_ + cc) * N_ + n] = acc[i][j][r] + b_out[cc];
      }
}

// ---------------------------------------------------------------------------
extern "C" void kernel_launch(void* const* d_in, const int* in_sizes, int n_in,
                              void* d_out, int out_size, void* d_ws, size_t ws_size,
                              hipStream_t stream) {
  (void)in_sizes; (void)n_in; (void)out_size; (void)ws_size;
  const float* x     = (const float*)d_in[0];
  const float* w_qkv = (const float*)d_in[1];
  const float* w_out = (const float*)d_in[2];
  const float* b_out = (const float*)d_in[3];
  float* out = (float*)d_out;

  const size_t qkv_elems = (size_t)B_ * HEADS_ * N_ * DH_;        // 4M halves
  char* ws = (char*)d_ws;
  _Float16* Q   = (_Float16*)(ws);
  _Float16* K   = (_Float16*)(ws + 2 * qkv_elems);
  _Float16* V   = (_Float16*)(ws + 4 * qkv_elems);
  _Float16* att = (_Float16*)(ws + 6 * qkv_elems);                // 32 MB total

  dim3 g1(QKVROWS / G_TM, N_ / G_TN, B_);          // 12 x 32 x 2
  qkv_proj_wmma<<<g1, 256, 0, stream>>>(x, w_qkv, Q, K, V);

  dim3 g2(N_ / 128, B_ * HEADS_);                  // 32 x 16
  flash_attn_wmma<<<g2, 256, 0, stream>>>(Q, K, V, att);

  dim3 g3(C_ / G_TM, N_ / G_TN, B_);               // 2 x 32 x 2
  out_proj_wmma<<<g3, 256, 0, stream>>>(w_out, att, b_out, out);
}
